// Memory_1022202217298
// MI455X (gfx1250) — matrix operand order repaired and checked by reading, not compiled, not run on hardware
//
#include <hip/hip_runtime.h>
#include <hip/hip_bf16.h>

// ---------------------------------------------------------------------------
// Cosine-similarity key-value memory (Kaiser-style) for MI455X / gfx1250.
//   q  = l2norm(input)  [B,D]     (bf16)
//   kn = l2norm(keys)   [M,D]     (bf16)
//   sims = q @ kn^T               (v_wmma_f32_16x16x32_bf16, f32 accum)
//   out[b] = sum_k softmax(40*top256(sims[b]))_k * value[idx_k]
//
// HBM traffic plan: keys read once (256 MB f32) for normalization; kn (134 MB
// bf16) written once and — because consecutive blocks share a kn chunk and
// MI455X has a 192 MB L2 — re-read from HBM ~once across all 64 row tiles.
// ---------------------------------------------------------------------------

typedef __attribute__((ext_vector_type(16))) __bf16 v16bf;
typedef __attribute__((ext_vector_type(8)))  __bf16 v8bf;
typedef __attribute__((ext_vector_type(8)))  float  v8f;

constexpr int D_DIM  = 256;
constexpr int B_ROWS = 1024;
constexpr int M_KEYS = 262144;
constexpr int K_TOP  = 256;
constexpr int MC     = 4096;              // columns per chunk (stage-1 tile)
constexpr int NCHUNK = M_KEYS / MC;       // 64
constexpr int NBIN   = 256;               // histogram bins over sims in [-1,1]
constexpr float INV_TEMP = 40.0f;

__device__ __forceinline__ int bin_of(float s) {
  float t = (s + 1.0f) * 128.0f;
  t = fminf(255.0f, fmaxf(0.0f, t));
  return (int)t;
}

// ---------------------------------------------------------------------------
// Kernel 1: row-wise L2 normalize f32 -> bf16.  One wave (32 lanes) per row.
// ---------------------------------------------------------------------------
__global__ void __launch_bounds__(256)
l2norm_bf16_kernel(const float* __restrict__ src,
                   __bf16* __restrict__ dst, int nrows) {
  const int row  = blockIdx.x * 8 + (threadIdx.x >> 5);
  const int lane = threadIdx.x & 31;
  if (row >= nrows) return;
  const float* p = src + (size_t)row * D_DIM + lane * 8;
  float4 a = *(const float4*)p;
  float4 b = *(const float4*)(p + 4);
  float ss = a.x*a.x + a.y*a.y + a.z*a.z + a.w*a.w
           + b.x*b.x + b.y*b.y + b.z*b.z + b.w*b.w;
  #pragma unroll
  for (int off = 16; off > 0; off >>= 1) ss += __shfl_xor(ss, off, 32);
  const float r = rsqrtf(ss + 1e-8f);
  v8bf o;
  o[0] = (__bf16)(a.x * r); o[1] = (__bf16)(a.y * r);
  o[2] = (__bf16)(a.z * r); o[3] = (__bf16)(a.w * r);
  o[4] = (__bf16)(b.x * r); o[5] = (__bf16)(b.y * r);
  o[6] = (__bf16)(b.z * r); o[7] = (__bf16)(b.w * r);
  *(v8bf*)(dst + (size_t)row * D_DIM + lane * 8) = o;
}

// ---------------------------------------------------------------------------
// Kernel 2 (stage 1): fused WMMA GEMM + per-chunk exact top-256 select.
//   grid = (B_ROWS/16, NCHUNK)  -- x = row tile so consecutive blocks share a
//   kn chunk (2 MB) and hit L2; block = 256 threads = 8 waves.
//
// WMMA fragment layouts used here (16-bit, wave32; A/B use contiguous k, not the ISA table order):
//   A (16x32, M x K): lane l holds row M=(l&15); kh=(l>>4);
//     v16 element e <-> K = kb*32 + kh*16 + e            (32B contiguous)
//   B (32x16, K x N): lane l holds col N=(l&15); kh=(l>>4);
//     v16 element e <-> K = kb*32 + kh*16 + e            (32B contiguous)
//   C/D (16x16 f32): VGPR j, lane l -> row M = j + 8*(l>>4), col N = (l&15)
// ---------------------------------------------------------------------------
__global__ void __launch_bounds__(256)
gemm_chunk_topk_kernel(const __bf16* __restrict__ qn,
                       const __bf16* __restrict__ kn,
                       float* __restrict__ c_sim,
                       int*   __restrict__ c_idx) {
  extern __shared__ char smem_raw[];
  // qtile (8 KB) aliases the head of the sims region: it is consumed into
  // registers (afrag) before the GEMM's first sims store, with barriers
  // separating the phases.
  __bf16* qtile = (__bf16*)smem_raw;                      // [16][D_DIM] bf16
  float*  sims  = (float*)smem_raw;                       // [16][MC]  f32
  int*    hist  = (int*)(smem_raw + (size_t)16 * MC * 4); // [16][NBIN]
  int*    cnt   = hist + 16 * NBIN;                       // [16] emit slots
  int*    takeT = cnt + 16;                               // [16] tie takers
  int*    Tb    = takeT + 16;                             // [16] threshold bin
  int*    Rn    = Tb + 16;                                // [16] need from bin

  const int tid   = threadIdx.x;
  const int lane  = tid & 31;
  const int wv    = tid >> 5;
  const int khalf = lane >> 4;
  const int l15   = lane & 15;
  const int rowbase = blockIdx.x * 16;   // row tile   (x: fast-varying)
  const int chunk   = blockIdx.y;        // key chunk  (y: L2-resident reuse)

  for (int i = tid; i < 16 * NBIN; i += 256) hist[i] = 0;
  if (tid < 16) { cnt[tid] = 0; takeT[tid] = 0; Tb[tid] = 0; Rn[tid] = K_TOP; }

  // ---- async-stage the 16x256 q row tile (8 KB contiguous) into LDS -------
  // 256 threads x 16B x 2 shots; ASYNCcnt tracked, drained before use.
  {
    const unsigned long long gsrc =
        (unsigned long long)(const void*)(qn + (size_t)rowbase * D_DIM);
    #pragma unroll
    for (int shot = 0; shot < 2; ++shot) {
      const unsigned       ldsOff = (unsigned)(unsigned long long)(const void*)qtile
                                  + (unsigned)(shot * 4096 + tid * 16);
      const unsigned long long ga = gsrc + (unsigned long long)(shot * 4096 + tid * 16);
      asm volatile("global_load_async_to_lds_b128 %0, %1, off"
                   :: "v"(ldsOff), "v"(ga) : "memory");
    }
    asm volatile("s_wait_asynccnt 0x0" ::: "memory");
  }
  __syncthreads();

  // ---- build all 8 A fragments (16 rows x K=256) from LDS -----------------
  const __bf16* arow = qtile + (size_t)l15 * D_DIM;
  v16bf afrag[8];
  #pragma unroll
  for (int kb = 0; kb < 8; ++kb) {
    v8bf lo = *(const v8bf*)(arow + kb * 32 + khalf * 16);
    v8bf hi = *(const v8bf*)(arow + kb * 32 + khalf * 16 + 8);
    #pragma unroll
    for (int i = 0; i < 8; ++i) { afrag[kb][i] = lo[i]; afrag[kb][i + 8] = hi[i]; }
  }
  __syncthreads();   // afrags in regs; sims region may now be overwritten

  // ---- WMMA: each wave runs TWO independent column tiles per iteration ----
  // (two accumulator chains + double-buffered B frags -> loads for kb+1 are
  //  in flight while kb's WMMAs execute; no per-WMMA loadcnt-0 stall)
  const __bf16* knc = kn + (size_t)chunk * MC * D_DIM;
  for (int t = wv; t < MC / 32; t += 8) {
    const int t2 = t + MC / 32;
    const __bf16* brow1 = knc + (size_t)(t  * 16 + l15) * D_DIM + khalf * 16;
    const __bf16* brow2 = knc + (size_t)(t2 * 16 + l15) * D_DIM + khalf * 16;
    __builtin_prefetch(brow1 + 8 * 16 * D_DIM, 0, 0);  // next iter's tile 1
    __builtin_prefetch(brow2 + 8 * 16 * D_DIM, 0, 0);  // next iter's tile 2

    v8f acc1 = {}, acc2 = {};
    v16bf b1 = *(const v16bf*)(brow1);
    v16bf b2 = *(const v16bf*)(brow2);
    #pragma unroll
    for (int kb = 0; kb < 8; ++kb) {
      v16bf nb1 = b1, nb2 = b2;
      if (kb < 7) {
        nb1 = *(const v16bf*)(brow1 + (kb + 1) * 32);
        nb2 = *(const v16bf*)(brow2 + (kb + 1) * 32);
      }
      acc1 = __builtin_amdgcn_wmma_f32_16x16x32_bf16(
          false, afrag[kb], false, b1, (short)0, acc1, false, false);
      acc2 = __builtin_amdgcn_wmma_f32_16x16x32_bf16(
          false, afrag[kb], false, b2, (short)0, acc2, false, false);
      b1 = nb1; b2 = nb2;
    }
    #pragma unroll
    for (int j = 0; j < 8; ++j) {
      sims[(j + 8 * khalf) * MC + t  * 16 + l15] = acc1[j];
      sims[(j + 8 * khalf) * MC + t2 * 16 + l15] = acc2[j];
    }
  }
  __syncthreads();

  // ---- per-row histogram (16 threads per row, contiguous strips) ----------
  const int r   = tid >> 4;
  const int sub = tid & 15;
  const int seg = MC / 16;  // 256 values per thread
  for (int c = sub * seg; c < sub * seg + seg; ++c)
    atomicAdd(&hist[r * NBIN + bin_of(sims[r * MC + c])], 1);
  __syncthreads();

  // ---- per-row threshold: largest bin T s.t. count(bin >= T) >= K_TOP ----
  if (tid < 16) {
    int cum = 0, T = 0, R = K_TOP;
    for (int bb = NBIN - 1; bb >= 0; --bb) {
      int h = hist[tid * NBIN + bb];
      if (cum + h >= K_TOP) { T = bb; R = K_TOP - cum; break; }
      cum += h;
    }
    Tb[tid] = T; Rn[tid] = R;
  }
  __syncthreads();

  // ---- emit exactly K_TOP candidates per row for this chunk ---------------
  const int T  = Tb[r];
  const int R  = Rn[r];
  const int gr = rowbase + r;
  float* osim = c_sim + ((size_t)gr * NCHUNK + chunk) * K_TOP;
  int*   oidx = c_idx + ((size_t)gr * NCHUNK + chunk) * K_TOP;
  for (int c = sub * seg; c < sub * seg + seg; ++c) {
    const float s = sims[r * MC + c];
    const int bb = bin_of(s);
    bool take = (bb > T);
    if (bb == T) take = (atomicAdd(&takeT[r], 1) < R);
    if (take) {
      const int slot = atomicAdd(&cnt[r], 1);  // < K_TOP by construction
      osim[slot] = s;
      oidx[slot] = chunk * MC + c;
    }
  }
}

// ---------------------------------------------------------------------------
// Kernel 3 (stage 2): per-row merge of 64*256 candidates -> exact top-256,
// softmax(40*s), weighted sum with value[].  One 256-thread block per row.
// ---------------------------------------------------------------------------
__global__ void __launch_bounds__(256)
topk_softmax_kernel(const float* __restrict__ c_sim,
                    const int*   __restrict__ c_idx,
                    const float* __restrict__ value,
                    float* __restrict__ out) {
  __shared__ int hist[NBIN];
  __shared__ int takeT;
  __shared__ int TR[2];
  __shared__ float rnum[8], rden[8];

  const int tid  = threadIdx.x;
  const int lane = tid & 31;
  const int wv   = tid >> 5;
  const int row  = blockIdx.x;
  const int NC   = NCHUNK * K_TOP;  // 16384 candidates
  const float* s_base = c_sim + (size_t)row * NC;
  const int*   i_base = c_idx + (size_t)row * NC;

  for (int i = tid; i < NBIN; i += 256) hist[i] = 0;
  if (tid == 0) takeT = 0;
  __syncthreads();

  for (int j = tid; j < NC; j += 256) atomicAdd(&hist[bin_of(s_base[j])], 1);
  __syncthreads();

  if (tid == 0) {
    int cum = 0, T = 0, R = K_TOP;
    for (int bb = NBIN - 1; bb >= 0; --bb) {
      int h = hist[bb];
      if (cum + h >= K_TOP) { T = bb; R = K_TOP - cum; break; }
      cum += h;
    }
    TR[0] = T; TR[1] = R;
  }
  __syncthreads();
  const int T = TR[0], R = TR[1];

  // exp(40*s) is safe in f32 for s in [-1,1]; num/den ratio cancels scale.
  float num = 0.0f, den = 0.0f;
  for (int j = tid; j < NC; j += 256) {
    const float s = s_base[j];
    const int bb = bin_of(s);
    bool take = (bb > T);
    if (bb == T) take = (atomicAdd(&takeT, 1) < R);
    if (take) {
      const float w = __expf(INV_TEMP * s);
      num += w * value[i_base[j]];
      den += w;
    }
  }
  #pragma unroll
  for (int off = 16; off > 0; off >>= 1) {
    num += __shfl_xor(num, off, 32);
    den += __shfl_xor(den, off, 32);
  }
  if (lane == 0) { rnum[wv] = num; rden[wv] = den; }
  __syncthreads();
  if (tid == 0) {
    float N = 0.0f, Dn = 0.0f;
    #pragma unroll
    for (int i = 0; i < 8; ++i) { N += rnum[i]; Dn += rden[i]; }
    out[row] = N / Dn;
  }
}

// ---------------------------------------------------------------------------
// Host-side launcher.
// ---------------------------------------------------------------------------
extern "C" void kernel_launch(void* const* d_in, const int* in_sizes, int n_in,
                              void* d_out, int out_size, void* d_ws, size_t ws_size,
                              hipStream_t stream) {
  const float* inp   = (const float*)d_in[0];   // [B, D] f32
  const float* keys  = (const float*)d_in[1];   // [M, D] f32
  const float* value = (const float*)d_in[2];   // [M]    f32
  float* out = (float*)d_out;                   // [B]    f32

  char* ws = (char*)d_ws;
  size_t off = 0;
  auto align256 = [](size_t x) { return (x + 255) & ~(size_t)255; };

  __bf16* qn = (__bf16*)(ws + off);
  off = align256(off + (size_t)B_ROWS * D_DIM * 2);
  __bf16* kn = (__bf16*)(ws + off);
  off = align256(off + (size_t)M_KEYS * D_DIM * 2);
  float* c_sim = (float*)(ws + off);
  off = align256(off + (size_t)B_ROWS * NCHUNK * K_TOP * 4);
  int* c_idx = (int*)(ws + off);
  off = align256(off + (size_t)B_ROWS * NCHUNK * K_TOP * 4);
  (void)ws_size; (void)in_sizes; (void)n_in; (void)out_size;

  l2norm_bf16_kernel<<<B_ROWS / 8, 256, 0, stream>>>(inp, qn, B_ROWS);
  l2norm_bf16_kernel<<<M_KEYS / 8, 256, 0, stream>>>(keys, kn, M_KEYS);

  // x = row tile (fast-varying) so the 64 consecutive blocks reuse one
  // 2 MB kn chunk out of the 192 MB L2; kn streams from HBM ~once total.
  dim3 g1(B_ROWS / 16, NCHUNK);
  size_t smem = (size_t)16 * MC * 4 + (size_t)16 * NBIN * 4 + 4 * 16 * 4;
  gemm_chunk_topk_kernel<<<g1, 256, smem, stream>>>(qn, kn, c_sim, c_idx);

  topk_softmax_kernel<<<B_ROWS, 256, 0, stream>>>(c_sim, c_idx, value, out);
}
